// My_GATLayer_9981503996078
// MI455X (gfx1250) — matrix operand order repaired and checked by reading, not compiled, not run on hardware
//
#include <hip/hip_runtime.h>
#include <hip/hip_bf16.h>

#define D 64
#define ROWS_PER_BLOCK 64
#define ASYNC_STAGE 1   // stage weights into LDS with global_load_async_to_lds_b128

typedef __attribute__((ext_vector_type(16))) __bf16 v16bf;
typedef __attribute__((ext_vector_type(8)))  float  v8f;

union V16U { v16bf v; uint4 q[2]; };

__device__ __forceinline__ unsigned short bf16_bits(float f) {
    union { float f; unsigned u; } x; x.f = f;
    unsigned r = x.u + 0x7fffu + ((x.u >> 16) & 1u);   // round-to-nearest-even
    return (unsigned short)(r >> 16);
}
__device__ __forceinline__ unsigned pack2_bf16(const float* p) {
    return (unsigned)bf16_bits(p[0]) | ((unsigned)bf16_bits(p[1]) << 16);
}

// order-preserving float <-> uint encoding for atomicMax on floats
__device__ __forceinline__ unsigned ord_encode(float f) {
    int fi = __float_as_int(f);
    return (fi >= 0) ? ((unsigned)fi | 0x80000000u) : ~(unsigned)fi;
}
__device__ __forceinline__ float ord_decode(unsigned u) {
    int fi = (u & 0x80000000u) ? (int)(u ^ 0x80000000u) : (int)~u;
    return __int_as_float(fi);
}

// ---------------------------------------------------------------------------
// Kernel 0: pack h, Wself, Wfunc to bf16 (row-major) once per launch.
// ---------------------------------------------------------------------------
__global__ void gat_pack_kernel(const float* __restrict__ h, const float* __restrict__ Wself,
                                const float* __restrict__ Wfunc, unsigned* __restrict__ h16,
                                unsigned* __restrict__ ws16, unsigned* __restrict__ wf16,
                                int nh2, int nw2) {
    int i = blockIdx.x * blockDim.x + threadIdx.x;
    if (i < nh2) {
        h16[i] = pack2_bf16(h + 2 * (size_t)i);
    } else if (i < nh2 + nw2) {
        int j = i - nh2;
        ws16[j] = pack2_bf16(Wself + 2 * (size_t)j);
    } else if (i < nh2 + 2 * nw2) {
        int j = i - nh2 - nw2;
        wf16[j] = pack2_bf16(Wfunc + 2 * (size_t)j);
    }
}

// ---------------------------------------------------------------------------
// Kernel 1: init. Zero denom/deg/e_max (per node) and d_out (agg accumulator).
// ---------------------------------------------------------------------------
__global__ void gat_init_kernel(float* __restrict__ out_agg, unsigned* __restrict__ emax,
                                float* __restrict__ denom, float* __restrict__ deg, int n_nodes) {
    int i = blockIdx.x * blockDim.x + threadIdx.x;
    int total = n_nodes * D;
    if (i < total) out_agg[i] = 0.0f;
    if (i < n_nodes) { emax[i] = 0u; denom[i] = 0.0f; deg[i] = 0.0f; }
}

// ---------------------------------------------------------------------------
// Kernel 2: dual GEMM via WMMA bf16.  h_s = h @ Wself^T,  z = h @ Wfunc^T.
// 128 threads = 4 waves.  Block owns 64 rows (4 row-tiles); wave w owns
// columns [16w,16w+16).  Weights staged into LDS with async b128 loads once
// per block; each wave then pulls its B operands (2 K-chunks x 2 matrices)
// into registers ONCE and reuses them across all 4 row-tiles.
// Inner loop per (tile,kchunk): 2 global_load_b128 (A) + 2 v_wmma.
// ---------------------------------------------------------------------------
__global__ void gat_gemm_kernel(const unsigned short* __restrict__ h16,
                                const unsigned short* __restrict__ ws16,
                                const unsigned short* __restrict__ wf16,
                                float* __restrict__ h_s,
                                float* __restrict__ z,
                                int n_nodes) {
    __shared__ unsigned short smem[2 * D * D];   // 16KB: [0..4095]=Wself, [4096..]=Wfunc

    const int t    = threadIdx.x;
    const int lane = t & 31;
    const int wave = t >> 5;                     // 0..3 -> 16-col tile
    const int m0   = blockIdx.x * ROWS_PER_BLOCK;
    const int n0   = wave * 16;

    // ---- stage both weight matrices into LDS (1024 x 16B chunks) ----
#if ASYNC_STAGE
#pragma unroll
    for (int i = 0; i < 8; ++i) {
        int c = t + i * 128;                     // 0..1023
        const char* srcp = (c < 512) ? ((const char*)ws16 + (size_t)c * 16)
                                     : ((const char*)wf16 + (size_t)(c - 512) * 16);
        unsigned lds_off = (unsigned)(unsigned long long)(&smem[c * 8]);
        unsigned long long gaddr = (unsigned long long)srcp;
        asm volatile("global_load_async_to_lds_b128 %0, %1, off"
                     :: "v"(lds_off), "v"(gaddr) : "memory");
    }
    asm volatile("s_wait_asynccnt 0x0" ::: "memory");
#else
#pragma unroll
    for (int i = 0; i < 8; ++i) {
        int c = t + i * 128;
        const uint4* srcp = (c < 512) ? ((const uint4*)ws16 + c)
                                      : ((const uint4*)wf16 + (c - 512));
        ((uint4*)smem)[c] = *srcp;
    }
#endif
    __syncthreads();

    // B operand regs: lane l holds column n0+(l&15); low lanes K 0..15,
    // high lanes K 16..31 of each 32-wide K chunk (contiguous k order, same as A).
    const int colB  = n0 + (lane & 15);
    const int bKoff = (lane < 16) ? 0 : 16;
    V16U bs[2], bfm[2];
#pragma unroll
    for (int kc2 = 0; kc2 < 2; ++kc2) {
        const unsigned short* wsrow = smem + colB * D + kc2 * 32 + bKoff;
        bs[kc2].q[0]  = *(const uint4*)(wsrow);
        bs[kc2].q[1]  = *(const uint4*)(wsrow + 8);
        bfm[kc2].q[0] = *(const uint4*)(wsrow + D * D);
        bfm[kc2].q[1] = *(const uint4*)(wsrow + D * D + 8);
    }

    // A row pattern: lane l holds row (l&15) of the tile; low lanes carry K
    // 0..15, high lanes K 16..31 (matches B).
    const int aKoff  = (lane < 16) ? 0 : 16;
    const int colOut = n0 + (lane & 15);

#pragma unroll
    for (int tr = 0; tr < 4; ++tr) {
        const int mt = m0 + tr * 16;
        int rowA = mt + (lane & 15);
        if (rowA >= n_nodes) rowA = n_nodes - 1;        // clamp: EXEC stays all-ones
        const unsigned short* hrow = h16 + (size_t)rowA * D;

        v8f acc_s = {};
        v8f acc_f = {};
#pragma unroll
        for (int kc2 = 0; kc2 < 2; ++kc2) {
            V16U a;
            a.q[0] = *(const uint4*)(hrow + kc2 * 32 + aKoff);
            a.q[1] = *(const uint4*)(hrow + kc2 * 32 + aKoff + 8);
            acc_s = __builtin_amdgcn_wmma_f32_16x16x32_bf16(false, a.v, false, bs[kc2].v,
                                                            (short)0, acc_s, false, false);
            acc_f = __builtin_amdgcn_wmma_f32_16x16x32_bf16(false, a.v, false, bfm[kc2].v,
                                                            (short)0, acc_f, false, false);
        }

        // C/D layout: VGPR i -> row mt+i (lanes 0-15) / mt+i+8 (lanes 16-31).
        const int rBase = mt + ((lane < 16) ? 0 : 8);
        if (mt + 16 <= n_nodes) {                        // uniform fast path
#pragma unroll
            for (int i = 0; i < 8; ++i) {
                int r = rBase + i;
                h_s[(size_t)r * D + colOut] = acc_s[i];
                z[(size_t)r * D + colOut]   = acc_f[i];
            }
        } else {
#pragma unroll
            for (int i = 0; i < 8; ++i) {
                int r = rBase + i;
                if (r < n_nodes) {
                    h_s[(size_t)r * D + colOut] = acc_s[i];
                    z[(size_t)r * D + colOut]   = acc_f[i];
                }
            }
        }
    }
}

// ---------------------------------------------------------------------------
// Kernel 3: attention score GEMV.  s_src = z @ w_src, s_dst = z @ w_dst.
// ---------------------------------------------------------------------------
__global__ void gat_sdot_kernel(const float* __restrict__ z, const float* __restrict__ Watt,
                                float* __restrict__ s_src, float* __restrict__ s_dst, int n_nodes) {
    int n = blockIdx.x * blockDim.x + threadIdx.x;
    if (n >= n_nodes) return;
    const float* zr = z + (size_t)n * D;
    float as = 0.0f, ad = 0.0f;
#pragma unroll
    for (int k = 0; k < D; k += 4) {
        float4 zv = *(const float4*)(zr + k);
        float4 ws = *(const float4*)(Watt + k);
        float4 wd = *(const float4*)(Watt + D + k);
        as += zv.x * ws.x + zv.y * ws.y + zv.z * ws.z + zv.w * ws.w;
        ad += zv.x * wd.x + zv.y * wd.y + zv.z * wd.z + zv.w * wd.w;
    }
    s_src[n] = as;
    s_dst[n] = ad;
}

__device__ __forceinline__ float leaky01(float x) { return x > 0.0f ? x : 0.01f * x; }

// ---------------------------------------------------------------------------
// Kernel 4: per-edge score + segment max (order-preserving uint atomicMax).
// ---------------------------------------------------------------------------
__global__ void gat_edge_max_kernel(const int* __restrict__ src, const int* __restrict__ dst,
                                    const float* __restrict__ s_src, const float* __restrict__ s_dst,
                                    unsigned* __restrict__ emax, int n_edges) {
    int e = blockIdx.x * blockDim.x + threadIdx.x;
    if (e >= n_edges) return;
    float ev = leaky01(s_src[src[e]] + s_dst[dst[e]]);
    atomicMax(&emax[dst[e]], ord_encode(ev));
}

// ---------------------------------------------------------------------------
// Kernel 5: e_exp, segment denom, degree.
// ---------------------------------------------------------------------------
__global__ void gat_edge_exp_kernel(const int* __restrict__ src, const int* __restrict__ dst,
                                    const float* __restrict__ s_src, const float* __restrict__ s_dst,
                                    const unsigned* __restrict__ emax,
                                    float* __restrict__ wedge, float* __restrict__ denom,
                                    float* __restrict__ deg, int n_edges) {
    int e = blockIdx.x * blockDim.x + threadIdx.x;
    if (e >= n_edges) return;
    int d = dst[e];
    float ev = leaky01(s_src[src[e]] + s_dst[d]);
    float w  = __expf(ev - ord_decode(emax[d]));
    wedge[e] = w;
    atomicAdd(&denom[d], w);
    atomicAdd(&deg[d], 1.0f);
}

// ---------------------------------------------------------------------------
// Kernel 6: scatter-aggregate  agg[dst] += a * z[src].  One wave per edge,
// lane handles features k=lane and k=lane+32 (z/agg stay in the 192MB L2).
// ---------------------------------------------------------------------------
__global__ void gat_edge_agg_kernel(const int* __restrict__ src, const int* __restrict__ dst,
                                    const float* __restrict__ z, const float* __restrict__ wedge,
                                    const float* __restrict__ denom, float* __restrict__ out_agg,
                                    int n_edges) {
    size_t tid = (size_t)blockIdx.x * blockDim.x + threadIdx.x;
    int e = (int)(tid >> 5);
    if (e >= n_edges) return;
    int lane = (int)(tid & 31);
    int s = src[e], d = dst[e];
    float a = wedge[e] / denom[d];
    const float* zr = z + (size_t)s * D;
    float* ar = out_agg + (size_t)d * D;
    atomicAdd(&ar[lane],      a * zr[lane]);
    atomicAdd(&ar[lane + 32], a * zr[lane + 32]);
}

// ---------------------------------------------------------------------------
// Kernel 7: finalize.  out = h + relu(deg>0 ? h_s+agg : h).  agg already in out.
// ---------------------------------------------------------------------------
__global__ void gat_finalize_kernel(const float* __restrict__ h, const float* __restrict__ h_s,
                                    const float* __restrict__ deg, float* __restrict__ out,
                                    int n_nodes) {
    int i = blockIdx.x * blockDim.x + threadIdx.x;
    if (i >= n_nodes * D) return;
    int n = i >> 6;
    float hv  = h[i];
    float v   = (deg[n] > 0.0f) ? (h_s[i] + out[i]) : hv;
    v = fmaxf(v, 0.0f);
    out[i] = hv + v;
}

extern "C" void kernel_launch(void* const* d_in, const int* in_sizes, int n_in,
                              void* d_out, int out_size, void* d_ws, size_t ws_size,
                              hipStream_t stream) {
    const float* h     = (const float*)d_in[0];
    // d_in[1] = snorm_n (unused by the reference math)
    const int*   src   = (const int*)d_in[2];
    const int*   dst   = (const int*)d_in[3];
    const float* Wself = (const float*)d_in[4];
    const float* Wfunc = (const float*)d_in[5];
    const float* Watt  = (const float*)d_in[6];

    const int N = in_sizes[0] / D;
    const int E = in_sizes[2];

    // workspace layout
    char* ws = (char*)d_ws;
    size_t off = 0;
    auto alloc = [&](size_t bytes) { char* p = ws + off; off += (bytes + 255) & ~(size_t)255; return p; };
    float*    h_s   = (float*)   alloc((size_t)N * D * sizeof(float));
    float*    z     = (float*)   alloc((size_t)N * D * sizeof(float));
    unsigned* h16   = (unsigned*)alloc((size_t)N * D * sizeof(unsigned short));
    unsigned* ws16  = (unsigned*)alloc((size_t)D * D * sizeof(unsigned short));
    unsigned* wf16  = (unsigned*)alloc((size_t)D * D * sizeof(unsigned short));
    float*    s_src = (float*)   alloc((size_t)N * sizeof(float));
    float*    s_dst = (float*)   alloc((size_t)N * sizeof(float));
    unsigned* emax  = (unsigned*)alloc((size_t)N * sizeof(unsigned));
    float*    denom = (float*)   alloc((size_t)N * sizeof(float));
    float*    deg   = (float*)   alloc((size_t)N * sizeof(float));
    float*    wedge = (float*)   alloc((size_t)E * sizeof(float));

    float* out = (float*)d_out;   // doubles as agg accumulator

    const int T = 256;
    const int nh2 = N * D / 2;
    const int nw2 = D * D / 2;
    int gPack  = (nh2 + 2 * nw2 + T - 1) / T;
    int gElem  = (N * D + T - 1) / T;
    int gRows  = (N + ROWS_PER_BLOCK - 1) / ROWS_PER_BLOCK;
    int gNode  = (N + T - 1) / T;
    int gEdge  = (E + T - 1) / T;
    long long aggThreads = (long long)E * 32;
    int gAgg   = (int)((aggThreads + T - 1) / T);

    gat_pack_kernel<<<gPack, T, 0, stream>>>(h, Wself, Wfunc, h16, ws16, wf16, nh2, nw2);
    gat_init_kernel<<<gElem, T, 0, stream>>>(out, emax, denom, deg, N);
    gat_gemm_kernel<<<gRows, 128, 0, stream>>>((const unsigned short*)h16,
                                               (const unsigned short*)ws16,
                                               (const unsigned short*)wf16, h_s, z, N);
    gat_sdot_kernel<<<gNode, T, 0, stream>>>(z, Watt, s_src, s_dst, N);
    gat_edge_max_kernel<<<gEdge, T, 0, stream>>>(src, dst, s_src, s_dst, emax, E);
    gat_edge_exp_kernel<<<gEdge, T, 0, stream>>>(src, dst, s_src, s_dst, emax, wedge, denom, deg, E);
    gat_edge_agg_kernel<<<gAgg, T, 0, stream>>>(src, dst, z, wedge, denom, out, E);
    gat_finalize_kernel<<<gElem, T, 0, stream>>>(h, h_s, deg, out, N);
}
